// HardcodedKVMemoryBlock_44547400794260
// MI455X (gfx1250) — matrix operand order repaired and checked by reading, not compiled, not run on hardware
//
#include <hip/hip_runtime.h>
#include <math.h>

#define PI_F 3.14159265358979323846f

// B=2, L=2048, D=256, P=32  ->  flattened rows N = 4096
#define LSEQ 2048
#define DDIM 256
#define NROWS 4096

typedef __attribute__((ext_vector_type(16))) __bf16 bf16x16;
typedef __attribute__((ext_vector_type(8)))  __bf16 bf16x8;
typedef __attribute__((ext_vector_type(8)))  float  f32x8;

// ---------------------------------------------------------------------------
// WMMA operand lane layouts used here (wave32; A and B both use contiguous k order):
// A (16x32 bf16): lane m=lane&15, hi=lane>>4:
//   a[0..7] = row m, K=k0+hi*16+i ; a[8..15] = row m, K=k0+hi*16+8+i
// B (32x16 bf16): lane n=lane&15, hi=lane>>4 holds col n, K=k0+hi*16+i
// C/D (16x16 f32): vgpr i -> row (hi*8+i), col (lane&15)
// ---------------------------------------------------------------------------

static __device__ __forceinline__ bf16x16 join8(bf16x8 lo, bf16x8 hi) {
  return __builtin_shufflevector(lo, hi, 0, 1, 2, 3, 4, 5, 6, 7, 8, 9, 10, 11,
                                 12, 13, 14, 15);
}

// A operand from f32 (LDS) with convert
static __device__ __forceinline__ bf16x16 load_a_f32(const float* __restrict__ src,
                                                     int ld, int lane, int k0) {
  const int m = lane & 15, hi = lane >> 4;
  const float* p = src + m * ld + k0 + hi * 16;
  bf16x16 a;
#pragma unroll
  for (int i = 0; i < 8; ++i) a[i] = (__bf16)p[i];
#pragma unroll
  for (int i = 0; i < 8; ++i) a[8 + i] = (__bf16)p[8 + i];
  return a;
}

// A operand from bf16 memory: two aligned 16B vector loads
static __device__ __forceinline__ bf16x16 load_a_bf16(const __bf16* __restrict__ src,
                                                      int ld, int lane, int k0) {
  const int m = lane & 15, hi = lane >> 4;
  const __bf16* p = src + m * ld + k0 + hi * 16;
  return join8(*(const bf16x8*)p, *(const bf16x8*)(p + 8));
}

// B = M^T, M row-major bf16 (rows of M are columns of B): one 32B load per lane
static __device__ __forceinline__ bf16x16 load_bT_bf16(const __bf16* __restrict__ M,
                                                       int ld, int lane, int k0,
                                                       int n0) {
  const int n = lane & 15, hi = lane >> 4;
  return *(const bf16x16*)(M + (size_t)(n0 + n) * ld + k0 + hi * 16);
}

static __device__ __forceinline__ f32x8 wmma_bf16(bf16x16 a, bf16x16 b, f32x8 c) {
  return __builtin_amdgcn_wmma_f32_16x16x32_bf16(false, a, false, b, (short)0, c,
                                                 false, false);
}

// Parallel LayerNorm over a 16x256 LDS tile (256 threads, biased var, eps 1e-5)
static __device__ __forceinline__ void layernorm_tile(
    float* tile, float* red, float* red2, float* mu, float* is,
    const float* __restrict__ gamma, const float* __restrict__ beta, int tid) {
  const int r = tid >> 4, cc = (tid & 15) << 4;
  float s = 0.f, s2 = 0.f;
#pragma unroll
  for (int j = 0; j < 16; ++j) {
    const float v = tile[r * 256 + cc + j];
    s += v; s2 += v * v;
  }
  red[tid] = s; red2[tid] = s2;
  __syncthreads();
  if (tid < 16) {
    float S = 0.f, S2 = 0.f;
#pragma unroll
    for (int j = 0; j < 16; ++j) { S += red[tid * 16 + j]; S2 += red2[tid * 16 + j]; }
    const float m = S * (1.0f / 256.0f);
    mu[tid] = m;
    is[tid] = rsqrtf(S2 * (1.0f / 256.0f) - m * m + 1e-5f);
  }
  __syncthreads();
  for (int i = tid; i < 16 * 256; i += 256) {
    const int rr = i >> 8, d = i & 255;
    tile[i] = (tile[i] - mu[rr]) * is[rr] * gamma[d] + beta[d];
  }
  __syncthreads();
}

// ---------------------------------------------------------------------------
// Kernel 0: one-time weight pre-swizzle f32 -> bf16 in B-operand lane order.
// Layout: tile t = (n0/16)*(K/32) + (k0/32); elem = t*512 + lane*16 + i
// ---------------------------------------------------------------------------
__global__ __launch_bounds__(256) void k_prep(const float* __restrict__ W,
                                              __bf16* __restrict__ out, int K,
                                              int N) {
  const int total = K * N;
  const int ktiles = K >> 5;
  for (int idx = blockIdx.x * 256 + threadIdx.x; idx < total;
       idx += gridDim.x * 256) {
    const int t = idx >> 9, r = idx & 511;
    const int lane = r >> 4, i = r & 15;
    const int kt = t % ktiles, nt = t / ktiles;
    const int k = (kt << 5) + ((lane >> 4) << 4) + i;
    const int n = (nt << 4) + (lane & 15);
    out[idx] = (__bf16)W[k * N + n];
  }
}

// ---------------------------------------------------------------------------
// Kernel 1: projections + phasor encode.
// ---------------------------------------------------------------------------
__global__ __launch_bounds__(256) void k_encode(
    const float* __restrict__ x, const __bf16* __restrict__ keyWsw,
    const float* __restrict__ keyb, const __bf16* __restrict__ valWsw,
    const float* __restrict__ valb, __bf16* __restrict__ q,
    __bf16* __restrict__ kmat, __bf16* __restrict__ mvT) {
  __shared__ __align__(16) float xs[16 * 256];
  __shared__ __align__(16) float accv[16 * 256];
  __shared__ __align__(16) float accp[16 * 32];
  const int tid = threadIdx.x;
  const int row0 = blockIdx.x * 16;

  {  // stage x tile with 16B vector copies
    const float4* src = (const float4*)(x + (size_t)row0 * 256);
    float4* dst = (float4*)xs;
    for (int i = tid; i < 16 * 64; i += 256) dst[i] = src[i];
  }
  __syncthreads();

  const int wave = tid >> 5, lane = tid & 31;
  // 16 n-tiles of value projection (N=256) + 2 n-tiles of key projection (N=32)
  for (int nt = wave; nt < 18; nt += 8) {
    const bool iskey = nt >= 16;
    const __bf16* Wsw = iskey ? keyWsw : valWsw;
    const int ldacc = iskey ? 32 : 256;
    const int n0 = iskey ? (nt - 16) * 16 : nt * 16;
    const __bf16* bp = Wsw + ((size_t)((n0 >> 4) * 8) << 9) + (lane << 4);
    bf16x16 b = *(const bf16x16*)bp;  // ping-pong double buffer
    f32x8 c = {};
#pragma unroll
    for (int kk = 0; kk < 8; ++kk) {
      const bf16x16 bn = (kk < 7) ? *(const bf16x16*)(bp + ((kk + 1) << 9)) : b;
      bf16x16 a = load_a_f32(xs, 256, lane, kk << 5);
      c = wmma_bf16(a, b, c);
      b = bn;
    }
    float* acc = iskey ? accp : accv;
    const int m0 = (lane >> 4) * 8, nn = lane & 15;
#pragma unroll
    for (int i = 0; i < 8; ++i) acc[(m0 + i) * ldacc + n0 + nn] = c[i];
  }
  __syncthreads();

  // phasor epilogue: Q at row l, shifted K at row l+1 (row 0 zeroed)
  for (int idx = tid; idx < 16 * 32; idx += 256) {
    const int r = idx >> 5, p = idx & 31;
    const int g = row0 + r;
    const int l = g & (LSEQ - 1);
    const float ph = tanhf(accp[idx] + keyb[p]) * PI_F;
    const float cs = cosf(ph), sn = sinf(ph);
    q[g * 64 + p]      = (__bf16)cs;
    q[g * 64 + 32 + p] = (__bf16)sn;
    if (l + 1 < LSEQ) {
      kmat[(g + 1) * 64 + p]      = (__bf16)cs;
      kmat[(g + 1) * 64 + 32 + p] = (__bf16)sn;
    }
    if (l == 0) {
      kmat[g * 64 + p]      = (__bf16)0.f;
      kmat[g * 64 + 32 + p] = (__bf16)0.f;
    }
  }
  // masked value epilogue, written TRANSPOSED: one 32B contiguous store/thread
  {
    const int d = tid;  // 0..255 = output feature
    const float bias = valb[d];
    bf16x16 vals;
#pragma unroll
    for (int r = 0; r < 16; ++r) {
      const int g = row0 + r;
      const int l = g & (LSEQ - 1);
      const float v = accv[r * 256 + d] + bias;
      vals[r] = (l & 1) ? (__bf16)v : (__bf16)0.f;
    }
    *(bf16x16*)(mvT + (size_t)d * NROWS + row0) = vals;
  }
}

// ---------------------------------------------------------------------------
// Kernel 2: causal "attention": retrieved = causal(Q K^T) @ mv, then scaling.
// Grid 512: (q-tile) x (column half); heavy q-tiles scheduled first.
// ---------------------------------------------------------------------------
__global__ __launch_bounds__(256) void k_attn(
    const __bf16* __restrict__ q, const __bf16* __restrict__ kmat,
    const __bf16* __restrict__ mvT, float* __restrict__ retr) {
  __shared__ __align__(32) __bf16 ss[16 * 128];  // bf16 score chunk
  const int tid = threadIdx.x, wave = tid >> 5, lane = tid & 31;
  const int bid = blockIdx.x;        // 0..511
  const int half = bid & 1;          // which 128 columns of D
  const int pair = bid >> 1;         // 0..255
  const int batch = pair & 1;
  const int qt = 127 - (pair >> 1);  // heavy (deep) tiles first
  const int qrow0 = (batch << 11) + (qt << 4);  // flattened Q row
  const int krow0 = batch << 11;                // flattened KV base row

  // Q tile A-operands (K=64 -> two k-steps), hoisted out of the chunk loop
  const bf16x16 qa0 = load_a_bf16(q + (size_t)qrow0 * 64, 64, lane, 0);
  const bf16x16 qa1 = load_a_bf16(q + (size_t)qrow0 * 64, 64, lane, 32);

  f32x8 c0 = {};
  const int n0 = (half << 7) + (wave << 4);  // this wave's 16 output columns
  const int m0 = (lane >> 4) * 8, nn = lane & 15;
  const int nchunks = (qt >> 3) + 1;
  const __bf16* kbase = kmat + (size_t)krow0 * 64;

  // preload chunk 0's K-matrix tile for this wave
  bf16x16 kb0 = load_bT_bf16(kbase, 64, lane, 0, wave * 16);
  bf16x16 kb1 = load_bT_bf16(kbase, 64, lane, 32, wave * 16);

  for (int ch = 0; ch < nchunks; ++ch) {
    const int kvbase = krow0 + ch * 128;
    const int kt = ch * 8 + wave;  // wave-uniform -> EXEC stays full for WMMA

    // preload first S@V B-operand of this chunk (global, barrier-independent)
    const __bf16* mp = mvT + (size_t)(n0 + nn) * NROWS + kvbase + (m0 << 1);
    bf16x16 mb = *(const bf16x16*)mp;

    if (ch + 1 < nchunks)  // hint next chunk's B columns (global_prefetch_b8)
      __builtin_prefetch(mp + 128, 0, 1);

    // --- stage S(16x128) into LDS; wave w owns KV tile ch*8+w ---
    if (kt <= qt) {
      f32x8 s = {};
      s = wmma_bf16(qa0, kb0, s);
      s = wmma_bf16(qa1, kb1, s);
#pragma unroll
      for (int i = 0; i < 8; ++i) {
        const int m = m0 + i;
        float v = s[i];
        if (kt == qt && nn > m) v = 0.f;  // causal mask on diagonal tile
        ss[m * 128 + wave * 16 + nn] = (__bf16)v;
      }
    } else {
#pragma unroll
      for (int i = 0; i < 8; ++i) ss[(m0 + i) * 128 + wave * 16 + nn] = (__bf16)0.f;
    }
    __syncthreads();

    // preload next chunk's K-matrix tile while the S@V phase runs
    if (ch + 1 < nchunks) {
      kb0 = load_bT_bf16(kbase, 64, lane, 0, (kt + 8) * 16);
      kb1 = load_bT_bf16(kbase, 64, lane, 32, (kt + 8) * 16);
    }

    // --- retrieved tile += S(16x128) @ mv(128 x 16 cols); ping-pong B loads
#pragma unroll
    for (int kk = 0; kk < 4; ++kk) {
      const bf16x16 mbn = (kk < 3) ? *(const bf16x16*)(mp + ((kk + 1) << 5)) : mb;
      bf16x16 a = load_a_bf16(ss, 128, lane, kk << 5);
      c0 = wmma_bf16(a, mb, c0);
      mb = mbn;
    }
    __syncthreads();
  }

  // epilogue: retrieved /= sqrt(valid)*sqrt(P), valid = max(floor((l+1)/2),1)
#pragma unroll
  for (int i = 0; i < 8; ++i) {
    const int m = m0 + i;
    const int l = qt * 16 + m;
    int valid = (l + 1) >> 1;
    if (valid < 1) valid = 1;
    const float scale = rsqrtf((float)valid * 32.0f);
    retr[(size_t)(qrow0 + m) * 256 + n0 + nn] = c0[i] * scale;
  }
}

// ---------------------------------------------------------------------------
// Kernel 3: LN1 + Linear(256->512) + exact GELU
// ---------------------------------------------------------------------------
__global__ __launch_bounds__(256) void k_mlp1(
    const float* __restrict__ retr, const float* __restrict__ g1,
    const float* __restrict__ b1v, const __bf16* __restrict__ W1sw,
    const float* __restrict__ bias1, float* __restrict__ hmid) {
  __shared__ __align__(16) float rs[16 * 256];
  __shared__ float red[256], red2[256], mu[16], is[16];
  const int tid = threadIdx.x, wave = tid >> 5, lane = tid & 31;
  const int row0 = blockIdx.x * 16;

  {
    const float4* src = (const float4*)(retr + (size_t)row0 * 256);
    float4* dst = (float4*)rs;
    for (int i = tid; i < 16 * 64; i += 256) dst[i] = src[i];
  }
  __syncthreads();
  layernorm_tile(rs, red, red2, mu, is, g1, b1v, tid);

  for (int nt = wave; nt < 32; nt += 8) {
    const int n0 = nt * 16;
    const __bf16* bp = W1sw + ((size_t)((n0 >> 4) * 8) << 9) + (lane << 4);
    bf16x16 b = *(const bf16x16*)bp;
    f32x8 c = {};
#pragma unroll
    for (int kk = 0; kk < 8; ++kk) {
      const bf16x16 bn = (kk < 7) ? *(const bf16x16*)(bp + ((kk + 1) << 9)) : b;
      bf16x16 a = load_a_f32(rs, 256, lane, kk << 5);
      c = wmma_bf16(a, b, c);
      b = bn;
    }
    const int m0 = (lane >> 4) * 8, nn = lane & 15;
#pragma unroll
    for (int i = 0; i < 8; ++i) {
      float v = c[i] + bias1[n0 + nn];
      v = 0.5f * v * (1.0f + erff(v * 0.70710678118654752f));  // exact GELU
      hmid[(size_t)(row0 + m0 + i) * 512 + n0 + nn] = v;
    }
  }
}

// ---------------------------------------------------------------------------
// Kernel 4: Linear(512->256)+b2 -> LN2 -> Linear(256->256)+bo -> + x (fused)
// ---------------------------------------------------------------------------
__global__ __launch_bounds__(256) void k_mlp2(
    const float* __restrict__ hmid, const __bf16* __restrict__ W2sw,
    const float* __restrict__ bias2, const float* __restrict__ g2,
    const float* __restrict__ b2v, const __bf16* __restrict__ Wosw,
    const float* __restrict__ bo, const float* __restrict__ x,
    float* __restrict__ out) {
  __shared__ __align__(16) float hs[16 * 512];
  __shared__ __align__(16) float rf[16 * 256];
  __shared__ float red[256], red2[256], mu[16], is[16];
  const int tid = threadIdx.x, wave = tid >> 5, lane = tid & 31;
  const int row0 = blockIdx.x * 16;
  const int n0 = wave * 32;
  const int m0 = (lane >> 4) * 8, nn = lane & 15;

  {
    const float4* src = (const float4*)(hmid + (size_t)row0 * 512);
    float4* dst = (float4*)hs;
    for (int i = tid; i < 16 * 128; i += 256) dst[i] = src[i];
  }
  __syncthreads();

  {  // refined = h @ W2 + b2  (into LDS); K=512 -> 16 k-tiles, ping-pong B
    const __bf16* bp0 = W2sw + ((size_t)((n0 >> 4) * 16) << 9) + (lane << 4);
    const __bf16* bp1 = W2sw + ((size_t)(((n0 + 16) >> 4) * 16) << 9) + (lane << 4);
    bf16x16 b0 = *(const bf16x16*)bp0;
    bf16x16 b1 = *(const bf16x16*)bp1;
    f32x8 c0 = {}, c1 = {};
#pragma unroll
    for (int kk = 0; kk < 16; ++kk) {
      bf16x16 bn0 = b0, bn1 = b1;
      if (kk < 15) {
        bn0 = *(const bf16x16*)(bp0 + ((kk + 1) << 9));
        bn1 = *(const bf16x16*)(bp1 + ((kk + 1) << 9));
      }
      bf16x16 a = load_a_f32(hs, 512, lane, kk << 5);
      c0 = wmma_bf16(a, b0, c0);
      c1 = wmma_bf16(a, b1, c1);
      b0 = bn0; b1 = bn1;
    }
#pragma unroll
    for (int i = 0; i < 8; ++i) {
      rf[(m0 + i) * 256 + n0 + nn]      = c0[i] + bias2[n0 + nn];
      rf[(m0 + i) * 256 + n0 + 16 + nn] = c1[i] + bias2[n0 + 16 + nn];
    }
  }
  __syncthreads();
  layernorm_tile(rf, red, red2, mu, is, g2, b2v, tid);

  {  // out = x + LN2(refined) @ Wo + bo ; K=256 -> 8 k-tiles, ping-pong B
    const __bf16* bp0 = Wosw + ((size_t)((n0 >> 4) * 8) << 9) + (lane << 4);
    const __bf16* bp1 = Wosw + ((size_t)(((n0 + 16) >> 4) * 8) << 9) + (lane << 4);
    bf16x16 b0 = *(const bf16x16*)bp0;
    bf16x16 b1 = *(const bf16x16*)bp1;
    f32x8 c0 = {}, c1 = {};
#pragma unroll
    for (int kk = 0; kk < 8; ++kk) {
      bf16x16 bn0 = b0, bn1 = b1;
      if (kk < 7) {
        bn0 = *(const bf16x16*)(bp0 + ((kk + 1) << 9));
        bn1 = *(const bf16x16*)(bp1 + ((kk + 1) << 9));
      }
      bf16x16 a = load_a_f32(rf, 256, lane, kk << 5);
      c0 = wmma_bf16(a, b0, c0);
      c1 = wmma_bf16(a, b1, c1);
      b0 = bn0; b1 = bn1;
    }
#pragma unroll
    for (int i = 0; i < 8; ++i) {
      const size_t gr = (size_t)(row0 + m0 + i) * 256;
      out[gr + n0 + nn]      = x[gr + n0 + nn] + c0[i] + bo[n0 + nn];
      out[gr + n0 + 16 + nn] = x[gr + n0 + 16 + nn] + c1[i] + bo[n0 + 16 + nn];
    }
  }
}

// ---------------------------------------------------------------------------
// Host launcher
// ---------------------------------------------------------------------------
extern "C" void kernel_launch(void* const* d_in, const int* in_sizes, int n_in,
                              void* d_out, int out_size, void* d_ws, size_t ws_size,
                              hipStream_t stream) {
  const float* x    = (const float*)d_in[0];
  const float* keyW = (const float*)d_in[1];
  const float* keyb = (const float*)d_in[2];
  const float* valW = (const float*)d_in[3];
  const float* valb = (const float*)d_in[4];
  const float* ln1g = (const float*)d_in[5];
  const float* ln1b = (const float*)d_in[6];
  const float* W1   = (const float*)d_in[7];
  const float* b1   = (const float*)d_in[8];
  const float* W2   = (const float*)d_in[9];
  const float* b2   = (const float*)d_in[10];
  const float* ln2g = (const float*)d_in[11];
  const float* ln2b = (const float*)d_in[12];
  const float* Wo   = (const float*)d_in[13];
  const float* bo   = (const float*)d_in[14];

  char* ws = (char*)d_ws;
  __bf16* q    = (__bf16*)(ws);                        // 4096*64  bf16 = 512 KB
  __bf16* kmat = (__bf16*)(ws + (512 << 10));          // 512 KB
  __bf16* mvT  = (__bf16*)(ws + (1024 << 10));         // 256 x 4096 bf16 = 2 MB
  float*  retr = (float*)(ws + (3072 << 10));          // 4096*256 f32  = 4 MB
  float*  hmid = (float*)(ws + (7168 << 10));          // 4096*512 f32  = 8 MB
  char*   wsw  = ws + (15360 << 10);                   // swizzled weights
  __bf16* keySw = (__bf16*)(wsw);                      //  16 KB (256x32)
  __bf16* valSw = (__bf16*)(wsw + 16384);              // 128 KB (256x256)
  __bf16* W1Sw  = (__bf16*)(wsw + 16384 + 131072);     // 256 KB (256x512)
  __bf16* W2Sw  = (__bf16*)(wsw + 16384 + 131072 + 262144);            // 256 KB
  __bf16* WoSw  = (__bf16*)(wsw + 16384 + 131072 + 262144 + 262144);   // 128 KB
  float*  out = (float*)d_out;

  k_prep<<<32, 256, 0, stream>>>(keyW, keySw, 256, 32);
  k_prep<<<256, 256, 0, stream>>>(valW, valSw, 256, 256);
  k_prep<<<512, 256, 0, stream>>>(W1, W1Sw, 256, 512);
  k_prep<<<512, 256, 0, stream>>>(W2, W2Sw, 512, 256);
  k_prep<<<256, 256, 0, stream>>>(Wo, WoSw, 256, 256);

  k_encode<<<NROWS / 16, 256, 0, stream>>>(x, keySw, keyb, valSw, valb, q, kmat, mvT);
  k_attn  <<<NROWS / 8, 256, 0, stream>>>(q, kmat, mvT, retr);
  k_mlp1  <<<NROWS / 16, 256, 0, stream>>>(retr, ln1g, ln1b, W1Sw, b1, hmid);
  k_mlp2  <<<NROWS / 16, 256, 0, stream>>>(hmid, W2Sw, b2, ln2g, ln2b, WoSw, bo, x, out);
}
